// HOTFormerStage_70265664962590
// MI455X (gfx1250) — matrix operand order repaired and checked by reading, not compiled, not run on hardware
//
#include <hip/hip_runtime.h>
#include <hip/hip_bf16.h>

// ---------------- problem constants ----------------
#define C_    256
#define H_    8
#define D_    32
#define K_    32
#define S_    33
#define NW_   8192
#define N_    (NW_ * K_)          // 262144 tokens
#define HID_  1024
#define MROWS (NW_ * S_)          // 270336 rows in x
#define SCALE_ 0.17677669529663687f // 32^-0.5

typedef __attribute__((ext_vector_type(16))) __bf16 bf16x16;
typedef __attribute__((ext_vector_type(8)))  float  f32x8;

// ---------------- helpers ----------------
__device__ __forceinline__ unsigned short f2bf(float f) {
    unsigned int u = __float_as_uint(f);
    u += 0x7FFFu + ((u >> 16) & 1u);      // round-to-nearest-even
    return (unsigned short)(u >> 16);
}
__device__ __forceinline__ float bf2f(unsigned short s) {
    return __uint_as_float(((unsigned int)s) << 16);
}
__device__ __forceinline__ float gelu_tanh(float x) {
    float x3 = x * x * x;
    return 0.5f * x * (1.f + tanhf(0.7978845608028654f * (x + 0.044715f * x3)));
}

// s_wait_asynccnt <= N (ASYNCcnt tracks global_load_async_to_lds_*)
template <int Ncnt>
__device__ __forceinline__ void wait_async() {
#if __has_builtin(__builtin_amdgcn_s_wait_asynccnt)
    __builtin_amdgcn_s_wait_asynccnt(Ncnt);
#else
    asm volatile("s_wait_asynccnt %0" :: "i"(Ncnt) : "memory");
#endif
}

// async copy of 16B from global (saddr + 32-bit voffset) into LDS at ldsoff
__device__ __forceinline__ void async_g2l_b128(unsigned ldsoff, unsigned voff,
                                               unsigned long long sbase) {
    asm volatile("global_load_async_to_lds_b128 %0, %1, %2"
                 :: "v"(ldsoff), "v"(voff), "s"(sbase)
                 : "memory");
}

// ---------------- kernel: build x = [relay ; window tokens + CPE] ----------------
__global__ __launch_bounds__(256) void build_x_kernel(
    const float* __restrict__ data, const float* __restrict__ rt,
    const float* __restrict__ cw,   // 3 x C (this block)
    const float* __restrict__ cb,   // C
    float* __restrict__ x)          // MROWS x C
{
    int row = blockIdx.x;       // 0 .. MROWS-1
    int c   = threadIdx.x;      // 0 .. 255
    int w = row / S_, s = row % S_;
    float val;
    if (s == 0) {
        val = rt[(size_t)w * C_ + c];
    } else {
        long t  = (long)w * K_ + (s - 1);
        float x0 = data[t * C_ + c];
        float xm = (t > 0)        ? data[(t - 1) * C_ + c] : 0.f;
        float xp = (t < N_ - 1)   ? data[(t + 1) * C_ + c] : 0.f;
        val = x0 + xm * cw[0 * C_ + c] + x0 * cw[1 * C_ + c] + xp * cw[2 * C_ + c] + cb[c];
    }
    x[(size_t)row * C_ + c] = val;
}

// ---------------- kernel: extract data / relay-tokens from x ----------------
__global__ __launch_bounds__(256) void extract_kernel(
    const float* __restrict__ x, float* __restrict__ data, float* __restrict__ rt)
{
    int row = blockIdx.x;
    int c   = threadIdx.x;
    int w = row / S_, s = row % S_;
    float v = x[(size_t)row * C_ + c];
    if (s == 0) rt[(size_t)w * C_ + c] = v;
    else        data[((size_t)w * K_ + (s - 1)) * C_ + c] = v;
}

// ---------------- kernel: layernorm (wave32 per row) -> bf16 ----------------
__global__ __launch_bounds__(256) void ln_kernel(
    const float* __restrict__ X, const float* __restrict__ g, const float* __restrict__ b,
    unsigned short* __restrict__ out)
{
    int row  = (int)((blockIdx.x * blockDim.x + threadIdx.x) >> 5);
    int lane = threadIdx.x & 31;
    if (row >= MROWS) return;
    const float* xr = X + (size_t)row * C_;
    float v[8];
    float s = 0.f;
#pragma unroll
    for (int j = 0; j < 8; ++j) { v[j] = xr[j * 32 + lane]; s += v[j]; }
#pragma unroll
    for (int off = 16; off; off >>= 1) s += __shfl_xor(s, off, 32);
    float mean = s * (1.f / 256.f);
    float var = 0.f;
#pragma unroll
    for (int j = 0; j < 8; ++j) { float d = v[j] - mean; var += d * d; }
#pragma unroll
    for (int off = 16; off; off >>= 1) var += __shfl_xor(var, off, 32);
    float r = rsqrtf(var * (1.f / 256.f) + 1e-5f);
#pragma unroll
    for (int j = 0; j < 8; ++j) {
        int c = j * 32 + lane;
        out[(size_t)row * C_ + c] = f2bf((v[j] - mean) * r * g[c] + b[c]);
    }
}

// ---------------- kernel: f32 weight (KxN) -> bf16 transposed (NxK) ----------------
__global__ __launch_bounds__(256) void convert_t_kernel(
    const float* __restrict__ W, unsigned short* __restrict__ Wt, int Kd, int Nd)
{
    int idx = blockIdx.x * blockDim.x + threadIdx.x;
    if (idx >= Kd * Nd) return;
    int n = idx / Kd, k = idx % Kd;
    Wt[idx] = f2bf(W[(size_t)k * Nd + n]);
}

// ---------------- WMMA GEMM: D = A(MxK,bf16) * Bt(NxK,bf16)^T + bias ----------------
// 256 threads = 8 waves; wave computes a 16x64 strip; block tile = 128 x 64.
// B tile (64 rows x 32 k, padded to 80B/row) staged in LDS with async
// global->LDS b128 copies, double-buffered (ASYNCcnt pipeline).
// EPI: 0 = bias -> bf16 store, 1 = gelu(bias+acc) -> bf16 store, 2 = X += gamma*(acc+bias)
#define BROW_BYTES 80            // 64 data bytes + 16 pad (bank spread)
#define BBUF_B128  (64 * BROW_BYTES / 16)   // uint4 slots per buffer (320)

template <int EPI>
__global__ __launch_bounds__(256) void gemm_bf16_kernel(
    const unsigned short* __restrict__ A,   // M x K
    const unsigned short* __restrict__ Bt,  // N x K (pre-transposed)
    const float* __restrict__ bias,         // N
    const float* __restrict__ gamma,        // N (EPI==2)
    float* __restrict__ X,                  // M x N residual stream (EPI==2)
    unsigned short* __restrict__ Obf,       // M x N bf16 out (EPI 0/1)
    int M, int N, int K)
{
    __shared__ uint4 smem4[2][BBUF_B128];   // 2 x 5120 B

    const int tid  = threadIdx.x;
    const int wave = tid >> 5;
    const int lane = tid & 31;
    const int half = lane >> 4;
    const int l16  = lane & 15;

    const int m0 = blockIdx.y * 128 + wave * 16;
    const int n0 = blockIdx.x * 64;
    const int mrow = m0 + l16;

    // async-copy assignment: thread t moves 16B chunk (row = t/4, chunk = t%4)
    const int brow = tid >> 2;   // 0..63
    const int bchk = tid & 3;    // 0..3
    const unsigned long long bbase = (unsigned long long)(const void*)Bt;
    const unsigned voff0 = (unsigned)((size_t)(n0 + brow) * K * 2) + (unsigned)(bchk * 16);
    const unsigned ldsdst0 = (unsigned)(size_t)(const void*)&smem4[0][0]
                           + (unsigned)(brow * BROW_BYTES + bchk * 16);

    const int nsteps = K >> 5;   // k-steps of 32

    // prologue: stage 0
    async_g2l_b128(ldsdst0, voff0, bbase);

    const unsigned int* Arow = (const unsigned int*)(A + (size_t)mrow * K);
    f32x8 acc[4] = {};

    for (int s = 0; s < nsteps; ++s) {
        const int kk = s * 32;

        // A fragment: 16x32 bf16, ISA layout (lanes<16: K 0-7,16-23 ; lanes>=16: +8)
        union { bf16x16 v; unsigned int u[8]; } af;
#pragma unroll
        for (int r = 0; r < 8; ++r) {
            int kb = ((r < 4) ? 0 : 16) + half * 8 + 2 * (r & 3);
            af.u[r] = Arow[(kk + kb) >> 1];
        }

        // issue next B stage while this one computes
        if (s + 1 < nsteps) {
            async_g2l_b128(ldsdst0 + (unsigned)(((s + 1) & 1) * 5120),
                           voff0 + (unsigned)((s + 1) * 64), bbase);
            wait_async<1>();     // stage s landed (in-order completion)
        } else {
            wait_async<0>();
        }
        __syncthreads();         // all waves' async writes visible

        const char* bb = (const char*)&smem4[s & 1][0];

        // preload all 4 B fragments (one ds clause), then 4 back-to-back WMMAs
        union { bf16x16 v; uint4 q[2]; } bfm[4];
#pragma unroll
        for (int j = 0; j < 4; ++j) {
            // B fragment: 32x16 bf16; lanes 0-15 hold K 0-7,16-23, lanes 16-31 K 8-15,24-31
            const uint4* bp = (const uint4*)(bb + (16 * j + l16) * BROW_BYTES + half * 16);
            bfm[j].q[0] = bp[0];
            bfm[j].q[1] = bp[2];
        }
#pragma unroll
        for (int j = 0; j < 4; ++j) {
            acc[j] = __builtin_amdgcn_wmma_f32_16x16x32_bf16(
                false, af.v, false, bfm[j].v, (short)0, acc[j], false, false);
        }
        __syncthreads();         // protect buffer before it is overwritten
    }

    // C/D layout: VGPR r, lanes<16 -> M=r, lanes>=16 -> M=8+r; N = lane%16
#pragma unroll
    for (int j = 0; j < 4; ++j) {
        int n = n0 + 16 * j + l16;
        float bv = bias[n];
#pragma unroll
        for (int r = 0; r < 8; ++r) {
            int m = m0 + r + half * 8;
            float v = acc[j][r] + bv;
            if (EPI == 0) {
                Obf[(size_t)m * N + n] = f2bf(v);
            } else if (EPI == 1) {
                Obf[(size_t)m * N + n] = f2bf(gelu_tanh(v));
            } else {
                X[(size_t)m * N + n] += gamma[n] * v;
            }
        }
    }
}

// ---------------- attention: one wave per (window, head) ----------------
__global__ __launch_bounds__(256) void attn_kernel(
    const unsigned short* __restrict__ qkv,  // MROWS x 768 bf16 (q|k|v each 256)
    const float* __restrict__ rpe,           // H x S x S (this block)
    unsigned short* __restrict__ o)          // MROWS x 256 bf16
{
    int gwave = (int)((blockIdx.x * blockDim.x + threadIdx.x) >> 5);
    int lane  = threadIdx.x & 31;            // lane == d index (D_ == 32)
    int w = gwave >> 3;
    int h = gwave & 7;
    if (w >= NW_) return;

    const size_t base = (size_t)w * S_ * 768;
    const int    hc   = h * D_ + lane;

    float kv[S_], vv[S_];
#pragma unroll
    for (int s = 0; s < S_; ++s) {
        kv[s] = bf2f(qkv[base + (size_t)s * 768 + 256 + hc]);
        vv[s] = bf2f(qkv[base + (size_t)s * 768 + 512 + hc]);
    }
    const float* rp = rpe + (size_t)h * S_ * S_;

    for (int si = 0; si < S_; ++si) {
        float q = bf2f(qkv[base + (size_t)si * 768 + hc]);
        float sc[S_];
        float mx = -1e30f;
#pragma unroll
        for (int sj = 0; sj < S_; ++sj) {
            float p = q * kv[sj];
#pragma unroll
            for (int off = 16; off; off >>= 1) p += __shfl_xor(p, off, 32);
            p = p * SCALE_ + rp[si * S_ + sj];
            sc[sj] = p;
            mx = fmaxf(mx, p);
        }
        float sum = 0.f;
#pragma unroll
        for (int sj = 0; sj < S_; ++sj) { sc[sj] = __expf(sc[sj] - mx); sum += sc[sj]; }
        float inv = 1.f / sum;
        float acc = 0.f;
#pragma unroll
        for (int sj = 0; sj < S_; ++sj) acc += sc[sj] * vv[sj];
        o[(size_t)(w * S_ + si) * C_ + hc] = f2bf(acc * inv);
    }
}

// ---------------- host side ----------------
extern "C" void kernel_launch(void* const* d_in, const int* in_sizes, int n_in,
                              void* d_out, int out_size, void* d_ws, size_t ws_size,
                              hipStream_t stream) {
    (void)in_sizes; (void)n_in; (void)out_size; (void)ws_size;

    const float* in_data  = (const float*)d_in[0];
    const float* in_rt    = (const float*)d_in[1];
    const float* cpe_w    = (const float*)d_in[2];
    const float* cpe_b    = (const float*)d_in[3];
    const float* ln1_g    = (const float*)d_in[4];
    const float* ln1_b    = (const float*)d_in[5];
    const float* qkv_w    = (const float*)d_in[6];
    const float* qkv_b    = (const float*)d_in[7];
    const float* rpe      = (const float*)d_in[8];
    const float* proj_w   = (const float*)d_in[9];
    const float* proj_b   = (const float*)d_in[10];
    const float* ln2_g    = (const float*)d_in[11];
    const float* ln2_b    = (const float*)d_in[12];
    const float* mlp_w1   = (const float*)d_in[13];
    const float* mlp_b1   = (const float*)d_in[14];
    const float* mlp_w2   = (const float*)d_in[15];
    const float* mlp_b2   = (const float*)d_in[16];
    const float* gamma1   = (const float*)d_in[17];
    const float* gamma2   = (const float*)d_in[18];

    // workspace carve-up
    char* ws = (char*)d_ws;
    float*          x    = (float*)ws;          ws += (size_t)MROWS * C_ * 4;   // 277 MB
    unsigned short* big  = (unsigned short*)ws; ws += (size_t)MROWS * HID_ * 2; // 554 MB (qkv / hid)
    unsigned short* hb   = (unsigned short*)ws; ws += (size_t)MROWS * C_ * 2;   // 138 MB (h / h2)
    unsigned short* ob   = (unsigned short*)ws; ws += (size_t)MROWS * C_ * 2;   // 138 MB (attn out)
    unsigned short* qkvT = (unsigned short*)ws; ws += (size_t)(3 * C_) * C_ * 2;
    unsigned short* projT= (unsigned short*)ws; ws += (size_t)C_ * C_ * 2;
    unsigned short* w1T  = (unsigned short*)ws; ws += (size_t)HID_ * C_ * 2;
    unsigned short* w2T  = (unsigned short*)ws; ws += (size_t)C_ * HID_ * 2;

    float* out_data = (float*)d_out;                    // N x C
    float* out_rt   = (float*)d_out + (size_t)N_ * C_; // NW x C

    const int tiles_m = MROWS / 128;                    // 2112

    for (int i = 0; i < 2; ++i) {
        const float* src_data = (i == 0) ? in_data : out_data;
        const float* src_rt   = (i == 0) ? in_rt   : out_rt;

        // x = [relay ; data + cpe]
        build_x_kernel<<<MROWS, 256, 0, stream>>>(
            src_data, src_rt, cpe_w + (size_t)i * 3 * C_, cpe_b + (size_t)i * C_, x);

        // h = LN1(x) -> bf16
        ln_kernel<<<MROWS / 8, 256, 0, stream>>>(
            x, ln1_g + (size_t)i * C_, ln1_b + (size_t)i * C_, hb);

        // qkv = h @ qkv_w + qkv_b   (WMMA bf16, N=768, K=256)
        convert_t_kernel<<<(3 * C_ * C_) / 256, 256, 0, stream>>>(
            qkv_w + (size_t)i * C_ * 3 * C_, qkvT, C_, 3 * C_);
        gemm_bf16_kernel<0><<<dim3(768 / 64, tiles_m), 256, 0, stream>>>(
            hb, qkvT, qkv_b + (size_t)i * 3 * C_, nullptr, nullptr, big,
            MROWS, 3 * C_, C_);

        // attention (VALU, wave per window-head)
        attn_kernel<<<(NW_ * H_) / 8, 256, 0, stream>>>(
            big, rpe + (size_t)i * H_ * S_ * S_, ob);

        // x += gamma1 * (o @ proj_w + proj_b)   (WMMA, N=256, K=256)
        convert_t_kernel<<<(C_ * C_) / 256, 256, 0, stream>>>(
            proj_w + (size_t)i * C_ * C_, projT, C_, C_);
        gemm_bf16_kernel<2><<<dim3(256 / 64, tiles_m), 256, 0, stream>>>(
            ob, projT, proj_b + (size_t)i * C_, gamma1 + (size_t)i * C_, x, nullptr,
            MROWS, C_, C_);

        // h2 = LN2(x) -> bf16
        ln_kernel<<<MROWS / 8, 256, 0, stream>>>(
            x, ln2_g + (size_t)i * C_, ln2_b + (size_t)i * C_, hb);

        // hid = gelu(h2 @ mlp_w1 + mlp_b1)   (WMMA, N=1024, K=256)
        convert_t_kernel<<<(C_ * HID_) / 256, 256, 0, stream>>>(
            mlp_w1 + (size_t)i * C_ * HID_, w1T, C_, HID_);
        gemm_bf16_kernel<1><<<dim3(HID_ / 64, tiles_m), 256, 0, stream>>>(
            hb, w1T, mlp_b1 + (size_t)i * HID_, nullptr, nullptr, big,
            MROWS, HID_, C_);

        // x += gamma2 * (hid @ mlp_w2 + mlp_b2)   (WMMA, N=256, K=1024)
        convert_t_kernel<<<(HID_ * C_) / 256, 256, 0, stream>>>(
            mlp_w2 + (size_t)i * HID_ * C_, w2T, HID_, C_);
        gemm_bf16_kernel<2><<<dim3(256 / 64, tiles_m), 256, 0, stream>>>(
            big, w2T, mlp_b2 + (size_t)i * C_, gamma2 + (size_t)i * C_, x, nullptr,
            MROWS, C_, HID_);

        // split x -> (data, rt); d_out doubles as inter-block buffer
        extract_kernel<<<MROWS, 256, 0, stream>>>(x, out_data, out_rt);
    }
}
